// MPNNModel_13649406067046
// MI455X (gfx1250) — compile-verified, matrix operand order checked by reading, not run on hardware
//
#include <hip/hip_runtime.h>
#include <hip/hip_bf16.h>

// ---------------- problem constants ----------------
#define NNODES  50000
#define NEDGES  400000
#define NGRAPHS 32
#define EMB     64
#define LAYERS  4
#define BN_EPS  1e-5f

typedef __attribute__((ext_vector_type(16))) _Float16 v16h;
typedef __attribute__((ext_vector_type(8)))  float    v8f;

union HU { unsigned u; _Float16 h[2]; };

// f32 hardware atomic add (global_atomic_add_f32)
__device__ __forceinline__ void atomAddF(float* p, float v) {
    unsafeAtomicAdd(p, v);
}

// ---------------------------------------------------------------------------
// WMMA fragment helpers (wave32, V_WMMA_F32_16X16X32_F16)
//
// 16-bit A-matrix 16x32 layout (ISA 7.12.2): lane L holds row M = L&15;
// lanes 0-15 carry K-halves {0..7,16..23}, lanes 16-31 carry {8..15,24..31};
// VGPR v<4 -> K = kb+2v(+1), v>=4 -> K = 16+kb+2(v-4)(+1), kb = (L&16)?8:0.
// B (32x16) uses the mirrored layout with N = L&15.  The dword pattern is
// contiguous, so each fragment lowers to two ds_load_b128 from LDS.
// ---------------------------------------------------------------------------
__device__ __forceinline__ v16h load_frag_lds(const _Float16* base, int lane,
                                              int k0, int stride) {
    int m  = lane & 15;
    int kb = (lane & 16) ? 8 : 0;
    union { v16h h; unsigned u[8]; } f;
    const unsigned* rp = (const unsigned*)(base + m * stride);
#pragma unroll
    for (int v = 0; v < 4; ++v)
        f.u[v] = rp[(k0 + kb + 2 * v) >> 1];
#pragma unroll
    for (int v = 0; v < 4; ++v)
        f.u[4 + v] = rp[(k0 + 16 + kb + 2 * v) >> 1];
    return f.h;
}

__device__ __forceinline__ v16h load_frag_packed(const _Float16* wp, int frag, int lane) {
    return *(const v16h*)(wp + ((size_t)frag << 9) + (lane << 4));
}

__device__ __forceinline__ v8f wmma_f16(v16h a, v16h b, v8f c) {
    return __builtin_amdgcn_wmma_f32_16x16x32_f16(false, a, false, b,
                                                  (short)0, c, false, false);
}

// ---------------------------------------------------------------------------
// Shared GEMM epilogue: add bias, accumulate BN statistics (LDS reduction),
// and write the 16x64 f16 output tile.
//  - fast path (whole tile in range, wave-uniform): accumulators -> LDS
//    staging tile -> coalesced global_store_b128 (64B/lane)
//  - TAIL instantiation adds a per-element guarded path (node kernels only)
// ---------------------------------------------------------------------------
template<bool TAIL>
__device__ __forceinline__ void tile_epilogue(
    v8f acc[4], const float* __restrict__ bias,
    _Float16* __restrict__ yout, long long r0, long long nRows,
    _Float16 (*outs)[64],            // this wave's 16x64 staging region
    float* sSum, float* sSqs, int lane)
{
    int rowoff = (lane & 16) ? 8 : 0;
    if (!TAIL || (r0 + 16 <= nRows)) {   // wave-uniform fast path
#pragma unroll
        for (int nt = 0; nt < 4; ++nt) {
            int n = nt * 16 + (lane & 15);
            float bv = bias[n];
            float ps = 0.f, pq = 0.f;
#pragma unroll
            for (int v = 0; v < 8; ++v) {
                float val = acc[nt][v] + bv;
                outs[rowoff + v][n] = (_Float16)val;
                ps += val; pq += val * val;
            }
            ps += __shfl_xor(ps, 16, 32);
            pq += __shfl_xor(pq, 16, 32);
            if (!(lane & 16)) { atomicAdd(&sSum[n], ps); atomicAdd(&sSqs[n], pq); }
        }
        __builtin_amdgcn_wave_barrier();   // keep ds stores before ds loads
        int r = lane & 15, half = lane >> 4;
        const uint4* sp = (const uint4*)(&outs[r][half * 32]);
        uint4* gp = (uint4*)(yout + (r0 + r) * 64 + half * 32);
#pragma unroll
        for (int i = 0; i < 4; ++i) gp[i] = sp[i];
    } else {                          // tail tile
#pragma unroll
        for (int nt = 0; nt < 4; ++nt) {
            int n = nt * 16 + (lane & 15);
            float bv = bias[n];
            float ps = 0.f, pq = 0.f;
#pragma unroll
            for (int v = 0; v < 8; ++v) {
                long long rr = r0 + rowoff + v;
                if (rr < nRows) {
                    float val = acc[nt][v] + bv;
                    yout[rr * 64 + n] = (_Float16)val;
                    ps += val; pq += val * val;
                }
            }
            ps += __shfl_xor(ps, 16, 32);
            pq += __shfl_xor(pq, 16, 32);
            if (!(lane & 16)) { atomicAdd(&sSum[n], ps); atomicAdd(&sSqs[n], pq); }
        }
    }
}

// ---------------------------------------------------------------------------
// Weight packing: W[lay][K][64] (f32, K zero-padded to Kpad) -> f16 fragments
// laid out [lay][frag = kt*4+nt][lane][slot] so a wave loads B with one
// contiguous 32B vector load per lane.
// ---------------------------------------------------------------------------
__global__ void k_pack_w(const float* __restrict__ W, _Float16* __restrict__ out,
                         int K, int Kpad, int nLay) {
    int tid = blockIdx.x * blockDim.x + threadIdx.x;
    int perLay = Kpad * 64;
    if (tid >= nLay * perLay) return;
    int lay  = tid / perLay;
    int idx  = tid - lay * perLay;
    int frag = idx >> 9;
    int lane = (idx >> 4) & 31;
    int s    = idx & 15;
    int kt = frag >> 2, nt = frag & 3;
    int v = s >> 1, j = s & 1;
    int kb = (lane & 16) ? 8 : 0;
    int kk = (v < 4) ? (kb + 2 * v + j) : (16 + kb + 2 * (v - 4) + j);
    int k  = kt * 32 + kk;
    int n  = nt * 16 + (lane & 15);
    float val = (k < K) ? W[(size_t)lay * K * 64 + (size_t)k * 64 + n] : 0.f;
    out[tid] = (_Float16)val;
}

__global__ void k_zero(float* __restrict__ p, long long n) {
    long long t = (long long)blockIdx.x * blockDim.x + threadIdx.x;
    if (t < n) p[t] = 0.f;
}

// ---------------------------------------------------------------------------
// Input linear: h = x(50000x11) @ W(11x64) + b ; emits f32 master + f16 copy
// ---------------------------------------------------------------------------
__global__ __launch_bounds__(128)
void k_lin_in(const float* __restrict__ x, const _Float16* __restrict__ wp,
              const float* __restrict__ bias, float* __restrict__ h32,
              _Float16* __restrict__ h16, long long nRows) {
    __shared__ _Float16 stage[4][16][32];
    int tid = threadIdx.x, wave = tid >> 5, lane = tid & 31;
    long long r0 = (long long)blockIdx.x * 64 + wave * 16;
    if (lane < 16) {
        long long row = r0 + lane;
        _Float16* op = &stage[wave][lane][0];
        if (row < nRows) {
            const float* ip = x + row * 11;
#pragma unroll
            for (int i = 0; i < 11; ++i) op[i] = (_Float16)ip[i];
#pragma unroll
            for (int i = 11; i < 32; ++i) op[i] = (_Float16)0.f;
        } else {
#pragma unroll
            for (int i = 0; i < 32; ++i) op[i] = (_Float16)0.f;
        }
    }
    __syncthreads();
    v8f acc[4] = {};
    v16h a = load_frag_lds(&stage[wave][0][0], lane, 0, 32);
#pragma unroll
    for (int nt = 0; nt < 4; ++nt)
        acc[nt] = wmma_f16(a, load_frag_packed(wp, nt, lane), acc[nt]);
    int rowoff = (lane & 16) ? 8 : 0;
    if (r0 + 16 <= nRows) {            // explicit wave-uniform fast path
#pragma unroll
        for (int nt = 0; nt < 4; ++nt) {
            int n = nt * 16 + (lane & 15);
            float bv = bias[n];
#pragma unroll
            for (int v = 0; v < 8; ++v) {
                long long row = r0 + rowoff + v;
                float val = acc[nt][v] + bv;
                h32[row * 64 + n] = val;
                h16[row * 64 + n] = (_Float16)val;
            }
        }
    } else {                           // tail tile
#pragma unroll
        for (int nt = 0; nt < 4; ++nt) {
            int n = nt * 16 + (lane & 15);
            float bv = bias[n];
#pragma unroll
            for (int v = 0; v < 8; ++v) {
                long long row = r0 + rowoff + v;
                if (row < nRows) {
                    float val = acc[nt][v] + bv;
                    h32[row * 64 + n] = val;
                    h16[row * 64 + n] = (_Float16)val;
                }
            }
        }
    }
}

// ---------------------------------------------------------------------------
// Edge MLP stage 1: m_in = [h[dst] | h[src] | edge_attr] (K=132 pad 160)
// y1 = m_in @ W1 + b1 (f16 out) + fused BN statistics.  Edge count is an
// exact multiple of 64, so the tail path is compiled out (TAIL=false).
// ---------------------------------------------------------------------------
__global__ __launch_bounds__(128)
void k_edge_gemm1(const _Float16* __restrict__ h16,
                  const float* __restrict__ eattr,
                  const int* __restrict__ srcIdx, const int* __restrict__ dstIdx,
                  const _Float16* __restrict__ wp, const float* __restrict__ bias,
                  _Float16* __restrict__ yout, float* __restrict__ gstats) {
    __shared__ _Float16 stage[4][16][160];
    __shared__ _Float16 outs[4][16][64];
    __shared__ float sSum[64], sSqs[64];
    int tid = threadIdx.x, wave = tid >> 5, lane = tid & 31;
    if (tid < 64) { sSum[tid] = 0.f; sSqs[tid] = 0.f; }
    long long e0 = (long long)blockIdx.x * 64 + wave * 16;
    int r = lane & 15, half = lane >> 4;
    long long e = e0 + r;
    int d = dstIdx[e], s = srcIdx[e];
    {   // gather the two 32-element halves of h[dst] and h[src]
        const uint4* pd = (const uint4*)(h16 + (size_t)d * 64 + half * 32);
        const uint4* ps = (const uint4*)(h16 + (size_t)s * 64 + half * 32);
        uint4* q0 = (uint4*)(&stage[wave][r][half * 32]);
        uint4* q1 = (uint4*)(&stage[wave][r][64 + half * 32]);
#pragma unroll
        for (int i = 0; i < 4; ++i) { q0[i] = pd[i]; q1[i] = ps[i]; }
    }
    if (half == 0) {
        const float* ea = eattr + e * 4;
#pragma unroll
        for (int i = 0; i < 4; ++i) stage[wave][r][128 + i] = (_Float16)ea[i];
    } else {
        unsigned* z = (unsigned*)(&stage[wave][r][132]);
#pragma unroll
        for (int i = 0; i < 14; ++i) z[i] = 0u;   // K pad 132..159
    }
    __syncthreads();
    v8f acc[4] = {};
#pragma unroll
    for (int kt = 0; kt < 5; ++kt) {
        v16h a = load_frag_lds(&stage[wave][0][0], lane, kt * 32, 160);
#pragma unroll
        for (int nt = 0; nt < 4; ++nt)
            acc[nt] = wmma_f16(a, load_frag_packed(wp, kt * 4 + nt, lane), acc[nt]);
    }
    tile_epilogue<false>(acc, bias, yout, e0, (long long)NEDGES, outs[wave], sSum, sSqs, lane);
    __syncthreads();
    if (tid < 64) { atomAddF(&gstats[tid], sSum[tid]); atomAddF(&gstats[64 + tid], sSqs[tid]); }
}

// ---------------------------------------------------------------------------
// Generic K=64 stage: y = relu(bn(xin)) @ W + b, fused BN stats on output.
// TAIL=false instantiation for edges (exact multiple of 64), TAIL=true for
// nodes.
// ---------------------------------------------------------------------------
template<bool TAIL>
__global__ __launch_bounds__(128)
void k_gemm_k64(const _Float16* __restrict__ xin, const float* __restrict__ bnp,
                const _Float16* __restrict__ wp, const float* __restrict__ bias,
                _Float16* __restrict__ yout, float* __restrict__ gstats,
                long long nRows) {
    __shared__ _Float16 stage[4][16][64];
    __shared__ _Float16 outs[4][16][64];
    __shared__ float sSum[64], sSqs[64];
    int tid = threadIdx.x, wave = tid >> 5, lane = tid & 31;
    if (tid < 64) { sSum[tid] = 0.f; sSqs[tid] = 0.f; }
    long long r0 = (long long)blockIdx.x * 64 + wave * 16;
    int r = lane & 15, half = lane >> 4;
    long long row = r0 + r;
    int c0 = half * 32;
    unsigned* op = (unsigned*)(&stage[wave][r][c0]);
    if (!TAIL || row < nRows) {
        const unsigned* ip = (const unsigned*)(xin + row * 64 + c0);
#pragma unroll
        for (int i = 0; i < 16; ++i) {
            HU in; in.u = ip[i];
            int n0 = c0 + 2 * i;
            float a0 = fmaxf((float)in.h[0] * bnp[n0]     + bnp[64 + n0],     0.f);
            float a1 = fmaxf((float)in.h[1] * bnp[n0 + 1] + bnp[64 + n0 + 1], 0.f);
            HU ou; ou.h[0] = (_Float16)a0; ou.h[1] = (_Float16)a1;
            op[i] = ou.u;
        }
    } else {
#pragma unroll
        for (int i = 0; i < 16; ++i) op[i] = 0u;
    }
    __syncthreads();
    v8f acc[4] = {};
#pragma unroll
    for (int kt = 0; kt < 2; ++kt) {
        v16h a = load_frag_lds(&stage[wave][0][0], lane, kt * 32, 64);
#pragma unroll
        for (int nt = 0; nt < 4; ++nt)
            acc[nt] = wmma_f16(a, load_frag_packed(wp, kt * 4 + nt, lane), acc[nt]);
    }
    tile_epilogue<TAIL>(acc, bias, yout, r0, nRows, outs[wave], sSum, sSqs, lane);
    __syncthreads();
    if (tid < 64) { atomAddF(&gstats[tid], sSum[tid]); atomAddF(&gstats[64 + tid], sSqs[tid]); }
}

// ---------------------------------------------------------------------------
// Node MLP stage 1: u_in = [h | aggr] (K=128, f32 -> f16 staged)
// ---------------------------------------------------------------------------
__global__ __launch_bounds__(128)
void k_node_gemm1(const float* __restrict__ h32, const float* __restrict__ aggr,
                  const _Float16* __restrict__ wp, const float* __restrict__ bias,
                  _Float16* __restrict__ yout, float* __restrict__ gstats,
                  long long nRows) {
    __shared__ _Float16 stage[4][16][128];
    __shared__ _Float16 outs[4][16][64];
    __shared__ float sSum[64], sSqs[64];
    int tid = threadIdx.x, wave = tid >> 5, lane = tid & 31;
    if (tid < 64) { sSum[tid] = 0.f; sSqs[tid] = 0.f; }
    long long r0 = (long long)blockIdx.x * 64 + wave * 16;
    int r = lane & 15, half = lane >> 4;
    long long row = r0 + r;
    unsigned* op = (unsigned*)(&stage[wave][r][half * 64]);
    if (row < nRows) {
        const float* sp = (half ? aggr : h32) + row * 64;
#pragma unroll
        for (int i = 0; i < 32; ++i) {
            float2 f = *(const float2*)(sp + 2 * i);
            HU u; u.h[0] = (_Float16)f.x; u.h[1] = (_Float16)f.y;
            op[i] = u.u;
        }
    } else {
#pragma unroll
        for (int i = 0; i < 32; ++i) op[i] = 0u;
    }
    __syncthreads();
    v8f acc[4] = {};
#pragma unroll
    for (int kt = 0; kt < 4; ++kt) {
        v16h a = load_frag_lds(&stage[wave][0][0], lane, kt * 32, 128);
#pragma unroll
        for (int nt = 0; nt < 4; ++nt)
            acc[nt] = wmma_f16(a, load_frag_packed(wp, kt * 4 + nt, lane), acc[nt]);
    }
    tile_epilogue<true>(acc, bias, yout, r0, nRows, outs[wave], sSum, sSqs, lane);
    __syncthreads();
    if (tid < 64) { atomAddF(&gstats[tid], sSum[tid]); atomAddF(&gstats[64 + tid], sSqs[tid]); }
}

// ---------------------------------------------------------------------------
// BN finalize: stats[0:64]=sum, stats[64:128]=sumsq -> scale/shift
// ---------------------------------------------------------------------------
__global__ void k_bn_fin(const float* __restrict__ stats, float cntInv,
                         const float* __restrict__ g, const float* __restrict__ be,
                         float* __restrict__ bnp) {
    int n = threadIdx.x;                 // 64 threads
    float mean = stats[n] * cntInv;
    float var  = stats[64 + n] * cntInv - mean * mean;
    float sc   = g[n] * rsqrtf(var + BN_EPS);
    bnp[n] = sc;
    bnp[64 + n] = be[n] - mean * sc;
}

// ---------------------------------------------------------------------------
// BN2 + ReLU + scatter-add into aggr (2 columns per thread, HW f32 atomics)
// ---------------------------------------------------------------------------
__global__ void k_scatter(const _Float16* __restrict__ y2, const float* __restrict__ bnp,
                          const int* __restrict__ dstIdx, float* __restrict__ aggr) {
    long long tid = (long long)blockIdx.x * blockDim.x + threadIdx.x;
    long long e = tid >> 5;
    int c = (int)(tid & 31) * 2;
    HU u; u.u = *(const unsigned*)(y2 + e * 64 + c);
    float a0 = fmaxf((float)u.h[0] * bnp[c]     + bnp[64 + c],     0.f);
    float a1 = fmaxf((float)u.h[1] * bnp[c + 1] + bnp[64 + c + 1], 0.f);
    long long base = (long long)dstIdx[e] * 64 + c;
    atomAddF(&aggr[base], a0);
    atomAddF(&aggr[base + 1], a1);
}

// h += relu(bn(z2)); refresh f16 copy
__global__ void k_residual(const _Float16* __restrict__ z2, const float* __restrict__ bnp,
                           float* __restrict__ h32, _Float16* __restrict__ h16) {
    long long tid = (long long)blockIdx.x * blockDim.x + threadIdx.x;
    int n = tid & 63;
    float u = fmaxf((float)z2[tid] * bnp[n] + bnp[64 + n], 0.f);
    float h = h32[tid] + u;
    h32[tid] = h;
    h16[tid] = (_Float16)h;
}

__global__ void k_pool(const float* __restrict__ h32, const int* __restrict__ batch,
                       float* __restrict__ gsum, float* __restrict__ gcnt) {
    long long tid = (long long)blockIdx.x * blockDim.x + threadIdx.x;
    long long i = tid >> 6; int n = (int)(tid & 63);
    int g = batch[i];
    atomAddF(&gsum[g * 64 + n], h32[tid]);
    if (n == 0) atomAddF(&gcnt[g], 1.f);
}

__global__ void k_pred(const float* __restrict__ gsum, const float* __restrict__ gcnt,
                       const float* __restrict__ pW, const float* __restrict__ pb,
                       float* __restrict__ out) {
    __shared__ float red[2];
    int g = blockIdx.x, t = threadIdx.x;    // 64 threads
    float cnt = fmaxf(gcnt[g], 1.f);
    float v = (gsum[g * 64 + t] / cnt) * pW[t];
#pragma unroll
    for (int off = 16; off; off >>= 1) v += __shfl_down(v, off, 32);
    if ((t & 31) == 0) red[t >> 5] = v;
    __syncthreads();
    if (t == 0) out[g] = red[0] + red[1] + pb[0];
}

// ---------------------------------------------------------------------------
extern "C" void kernel_launch(void* const* d_in, const int* in_sizes, int n_in,
                              void* d_out, int out_size, void* d_ws, size_t ws_size,
                              hipStream_t stream) {
    (void)in_sizes; (void)n_in; (void)out_size; (void)ws_size;
    const float* x      = (const float*)d_in[0];
    const float* eattr  = (const float*)d_in[1];
    const int*   eidx   = (const int*)d_in[2];
    const int*   batch  = (const int*)d_in[3];
    const float* linW   = (const float*)d_in[4];
    const float* linB   = (const float*)d_in[5];
    const float* msgW1  = (const float*)d_in[6];
    const float* msgB1  = (const float*)d_in[7];
    const float* msgG1  = (const float*)d_in[8];
    const float* msgBe1 = (const float*)d_in[9];
    const float* msgW2  = (const float*)d_in[10];
    const float* msgB2  = (const float*)d_in[11];
    const float* msgG2  = (const float*)d_in[12];
    const float* msgBe2 = (const float*)d_in[13];
    const float* updW1  = (const float*)d_in[14];
    const float* updB1  = (const float*)d_in[15];
    const float* updG1  = (const float*)d_in[16];
    const float* updBe1 = (const float*)d_in[17];
    const float* updW2  = (const float*)d_in[18];
    const float* updB2  = (const float*)d_in[19];
    const float* updG2  = (const float*)d_in[20];
    const float* updBe2 = (const float*)d_in[21];
    const float* predW  = (const float*)d_in[22];
    const float* predB  = (const float*)d_in[23];
    const int* srcIdx = eidx;            // edge_index[0] = j (source)
    const int* dstIdx = eidx + NEDGES;   // edge_index[1] = i (target)
    float* out = (float*)d_out;

    // ------- workspace carve-out (256B aligned slabs) -------
    char* ws = (char*)d_ws;
    size_t off = 0;
    auto alloc = [&](size_t b) -> char* {
        char* p = ws + off;
        off = (off + b + 255) & ~(size_t)255;
        return p;
    };
    float*    h32  = (float*)alloc((size_t)NNODES * 64 * 4);
    _Float16* h16f = (_Float16*)alloc((size_t)NNODES * 64 * 2);
    float*    aggr = (float*)alloc((size_t)NNODES * 64 * 4);
    _Float16* y1   = (_Float16*)alloc((size_t)NEDGES * 64 * 2);
    _Float16* y2   = (_Float16*)alloc((size_t)NEDGES * 64 * 2);
    _Float16* z1   = (_Float16*)alloc((size_t)NNODES * 64 * 2);
    _Float16* z2   = (_Float16*)alloc((size_t)NNODES * 64 * 2);
    _Float16* wpLin = (_Float16*)alloc((size_t)32 * 64 * 2);
    _Float16* wpM1  = (_Float16*)alloc((size_t)LAYERS * 160 * 64 * 2);
    _Float16* wpM2  = (_Float16*)alloc((size_t)LAYERS * 64 * 64 * 2);
    _Float16* wpU1  = (_Float16*)alloc((size_t)LAYERS * 128 * 64 * 2);
    _Float16* wpU2  = (_Float16*)alloc((size_t)LAYERS * 64 * 64 * 2);
    float* stats = (float*)alloc(16 * 128 * 4);
    float* bnp   = (float*)alloc(16 * 128 * 4);
    float* gsum  = (float*)alloc(32 * 64 * 4);   // 8192B -> gcnt is contiguous
    float* gcnt  = (float*)alloc(32 * 4);

    // ------- pack weights to WMMA B-fragment layout -------
    k_pack_w<<<(32 * 64 + 255) / 256, 256, 0, stream>>>(linW, wpLin, 11, 32, 1);
    k_pack_w<<<(LAYERS * 160 * 64 + 255) / 256, 256, 0, stream>>>(msgW1, wpM1, 132, 160, LAYERS);
    k_pack_w<<<(LAYERS * 64 * 64 + 255) / 256, 256, 0, stream>>>(msgW2, wpM2, 64, 64, LAYERS);
    k_pack_w<<<(LAYERS * 128 * 64 + 255) / 256, 256, 0, stream>>>(updW1, wpU1, 128, 128, LAYERS);
    k_pack_w<<<(LAYERS * 64 * 64 + 255) / 256, 256, 0, stream>>>(updW2, wpU2, 64, 64, LAYERS);
    // zero BN stat accumulators + pooling buffers (gsum/gcnt contiguous)
    k_zero<<<(16 * 128 + 255) / 256, 256, 0, stream>>>(stats, 16 * 128);
    k_zero<<<(32 * 64 + 32 + 255) / 256, 256, 0, stream>>>(gsum, 32 * 64 + 32);

    // ------- input linear -------
    k_lin_in<<<(NNODES + 63) / 64, 128, 0, stream>>>(x, wpLin, linB, h32, h16f, NNODES);

    const float invE = 1.f / (float)NEDGES;
    const float invN = 1.f / (float)NNODES;

    for (int l = 0; l < LAYERS; ++l) {
        float* st0 = stats + (l * 4 + 0) * 128; float* bp0 = bnp + (l * 4 + 0) * 128;
        float* st1 = stats + (l * 4 + 1) * 128; float* bp1 = bnp + (l * 4 + 1) * 128;
        float* st2 = stats + (l * 4 + 2) * 128; float* bp2 = bnp + (l * 4 + 2) * 128;
        float* st3 = stats + (l * 4 + 3) * 128; float* bp3 = bnp + (l * 4 + 3) * 128;

        k_zero<<<((long long)NNODES * 64 + 255) / 256, 256, 0, stream>>>(aggr, (long long)NNODES * 64);

        // edge MLP (400000 rows: exact multiple of 64 -> TAIL=false)
        k_edge_gemm1<<<NEDGES / 64, 128, 0, stream>>>(h16f, eattr, srcIdx, dstIdx,
                                                      wpM1 + (size_t)l * 160 * 64,
                                                      msgB1 + l * 64, y1, st0);
        k_bn_fin<<<1, 64, 0, stream>>>(st0, invE, msgG1 + l * 64, msgBe1 + l * 64, bp0);
        k_gemm_k64<false><<<NEDGES / 64, 128, 0, stream>>>(y1, bp0, wpM2 + (size_t)l * 64 * 64,
                                                           msgB2 + l * 64, y2, st1, (long long)NEDGES);
        k_bn_fin<<<1, 64, 0, stream>>>(st1, invE, msgG2 + l * 64, msgBe2 + l * 64, bp1);
        // aggregate messages
        k_scatter<<<(int)(((long long)NEDGES * 32) / 256), 256, 0, stream>>>(y2, bp1, dstIdx, aggr);

        // node MLP + residual (50000 rows: has a tail tile -> TAIL=true)
        k_node_gemm1<<<(NNODES + 63) / 64, 128, 0, stream>>>(h32, aggr,
                                                             wpU1 + (size_t)l * 128 * 64,
                                                             updB1 + l * 64, z1, st2, (long long)NNODES);
        k_bn_fin<<<1, 64, 0, stream>>>(st2, invN, updG1 + l * 64, updBe1 + l * 64, bp2);
        k_gemm_k64<true><<<(NNODES + 63) / 64, 128, 0, stream>>>(z1, bp2, wpU2 + (size_t)l * 64 * 64,
                                                                 updB2 + l * 64, z2, st3, (long long)NNODES);
        k_bn_fin<<<1, 64, 0, stream>>>(st3, invN, updG2 + l * 64, updBe2 + l * 64, bp3);
        k_residual<<<(int)(((long long)NNODES * 64) / 256), 256, 0, stream>>>(z2, bp3, h32, h16f);
    }

    // ------- global mean pool + prediction -------
    k_pool<<<(int)(((long long)NNODES * 64) / 256), 256, 0, stream>>>(h32, batch, gsum, gcnt);
    k_pred<<<NGRAPHS, 64, 0, stream>>>(gsum, gcnt, predW, predB, out);
}
